// HGT_90890097918079
// MI455X (gfx1250) — matrix operand order repaired and checked by reading, not compiled, not run on hardware
//
#include <hip/hip_runtime.h>
#include <math.h>

// ---------------------------------------------------------------------------
// Problem constants (from the reference)
// ---------------------------------------------------------------------------
#define DD   256     // hidden dim
#define HH   4       // heads
#define DHD  64      // dim per head
#define NB   16      // graphs per batch
#define GW   1056    // padded width of pooled feature row (1040 -> 1056, K%32==0)
#define N_OP  30000
#define N_VAR 20000
#define N_DCT 2000
#define M_OPG 10000
#define E_OPOP  200000
#define E_OPVAR 100000
#define E_VAROP 100000
#define E_DCTE  20000

typedef __attribute__((ext_vector_type(16))) _Float16 v16h;
typedef __attribute__((ext_vector_type(8)))  float    v8f;

__device__ __forceinline__ float gelu_exact(float v) {
    return 0.5f * v * (1.f + erff(v * 0.70710678118654752440f));
}

// float atomic max via sign-aware integer atomics (correct when buffer is
// initialized to -inf): positive -> signed atomicMax, negative -> unsigned atomicMin.
__device__ __forceinline__ void atomicMaxF(float* addr, float v) {
    if (v >= 0.f) atomicMax((int*)addr, __float_as_int(v));
    else          atomicMin((unsigned int*)addr, __float_as_uint(v));
}

// load 16 floats strided by ld (K rows {0..7,16..23} from p) and pack to a v16h fragment
__device__ __forceinline__ v16h load_frag16(const float* __restrict__ p, int ld) {
    v16h v;
#pragma unroll
    for (int s = 0; s < 16; ++s) v[s] = (_Float16)p[(long)(s < 8 ? s : s + 8) * ld];
    return v;
}

// ---------------------------------------------------------------------------
// Fast WMMA GEMM:  C[M,N] = act(A[M,K] * B[K,N] + bias)
// REQUIRES: M % 16 == 0, N % 64 == 0, K % 32 == 0, 16B-aligned rows.
// One wave (32 threads) computes a 16x64 C tile: 4 accumulators, 4 WMMAs/K-step,
// A fragment reused 4x. All B fragments are materialized in distinct register
// sets BEFORE the 4 back-to-back WMMAs, so the WMMA->VALU WAR hazard slots are
// filled by the next iteration's loads/converts instead of v_nops.
// grid = (N/64, M/16, nbatch). Batched with element strides for per-head einsums.
// ---------------------------------------------------------------------------
__global__ __launch_bounds__(32) void wmma_gemm_kernel(
    const float* __restrict__ A, const float* __restrict__ Bw,
    const float* __restrict__ bias, float* __restrict__ C,
    int K, int lda, int ldb, int ldc,
    long aBatch, long bBatch, long cBatch, int act)
{
    const int lane  = threadIdx.x & 31;
    const int tileM = blockIdx.y * 16;
    const int tileN = blockIdx.x * 64;
    A  += (long)blockIdx.z * aBatch;
    Bw += (long)blockIdx.z * bBatch;
    C  += (long)blockIdx.z * cBatch;

    const int half = lane >> 4;   // K-half selector (ISA 16-bit A layout)
    const int r    = lane & 15;   // row within tile

    const float* arp = A + (long)(tileM + r) * lda;

    v8f acc0 = {}, acc1 = {}, acc2 = {}, acc3 = {};

    for (int k0 = 0; k0 < K; k0 += 32) {
        // A fragment: lane<16 -> K = k0+{0..7,16..23}; lane>=16 -> k0+{8..15,24..31}
        const int ka = k0 + half * 8;
        const float4 af0 = *(const float4*)(arp + ka);
        const float4 af1 = *(const float4*)(arp + ka + 4);
        const float4 af2 = *(const float4*)(arp + ka + 16);
        const float4 af3 = *(const float4*)(arp + ka + 20);
        v16h a;
        a[0]  = (_Float16)af0.x; a[1]  = (_Float16)af0.y;
        a[2]  = (_Float16)af0.z; a[3]  = (_Float16)af0.w;
        a[4]  = (_Float16)af1.x; a[5]  = (_Float16)af1.y;
        a[6]  = (_Float16)af1.z; a[7]  = (_Float16)af1.w;
        a[8]  = (_Float16)af2.x; a[9]  = (_Float16)af2.y;
        a[10] = (_Float16)af2.z; a[11] = (_Float16)af2.w;
        a[12] = (_Float16)af3.x; a[13] = (_Float16)af3.y;
        a[14] = (_Float16)af3.z; a[15] = (_Float16)af3.w;

        // B fragments: lane l holds N col s*16+(l&15), K rows as A; sub-tile s covers N cols s*16..s*16+15
        const float* brp = Bw + (long)(k0 + half * 8) * ldb + tileN + r;
        const v16h b0 = load_frag16(brp, ldb);
        const v16h b1 = load_frag16(brp + 16, ldb);
        const v16h b2 = load_frag16(brp + 32, ldb);
        const v16h b3 = load_frag16(brp + 48, ldb);

        // 4 WMMAs back-to-back: no D->A/B RAW between them (distinct accumulators)
        acc0 = __builtin_amdgcn_wmma_f32_16x16x32_f16(false, a, false, b0, (short)0, acc0, false, false);
        acc1 = __builtin_amdgcn_wmma_f32_16x16x32_f16(false, a, false, b1, (short)0, acc1, false, false);
        acc2 = __builtin_amdgcn_wmma_f32_16x16x32_f16(false, a, false, b2, (short)0, acc2, false, false);
        acc3 = __builtin_amdgcn_wmma_f32_16x16x32_f16(false, a, false, b3, (short)0, acc3, false, false);
    }

    // C layout: VGPR i -> row tileM + half*8 + i, col tileN + s*16 + (lane&15)
    v8f accs[4] = {acc0, acc1, acc2, acc3};
#pragma unroll
    for (int s = 0; s < 4; ++s) {
        const int col = tileN + s * 16 + r;
        const float bv = bias ? bias[col] : 0.f;
#pragma unroll
        for (int i = 0; i < 8; ++i) {
            const int row = tileM + half * 8 + i;
            float v = accs[s][i] + bv;
            if (act == 1)      v = gelu_exact(v);
            else if (act == 2) v = (v > 0.f) ? v : 0.1f * v;
            C[(long)row * ldc + col] = v;
        }
    }
}

// ---------------------------------------------------------------------------
// Elementwise / utility kernels
// ---------------------------------------------------------------------------
__global__ void fill_kernel(float* p, float v, long n) {
    long t = (long)blockIdx.x * blockDim.x + threadIdx.x;
    if (t < n) p[t] = v;
}

__global__ void copy_kernel(const float* __restrict__ s, float* __restrict__ d, long n) {
    long t = (long)blockIdx.x * blockDim.x + threadIdx.x;
    if (t < n) d[t] = s[t];
}

__global__ void gather_rows_kernel(const float* __restrict__ src,
                                   const int* __restrict__ idx,
                                   float* __restrict__ dst, int nrows) {
    long t = (long)blockIdx.x * blockDim.x + threadIdx.x;
    if (t >= (long)nrows * DD) return;
    int row = (int)(t >> 8), c = (int)(t & 255);
    dst[t] = src[(long)idx[row] * DD + c];
}

__global__ void scatter_rows_kernel(const float* __restrict__ src,
                                    const int* __restrict__ idx,
                                    float* __restrict__ dst, int nrows) {
    long t = (long)blockIdx.x * blockDim.x + threadIdx.x;
    if (t >= (long)nrows * DD) return;
    int row = (int)(t >> 8), c = (int)(t & 255);
    dst[(long)idx[row] * DD + c] = src[t];
}

__global__ void gelu_kernel(float* x, long n) {
    long t = (long)blockIdx.x * blockDim.x + threadIdx.x;
    if (t < n) x[t] = gelu_exact(x[t]);
}

// out = sigmoid(skip)*o + (1-sigmoid(skip))*xin
__global__ void blend_kernel(const float* __restrict__ o, const float* __restrict__ xin,
                             const float* __restrict__ skip, float* __restrict__ xout, long n) {
    long t = (long)blockIdx.x * blockDim.x + threadIdx.x;
    if (t >= n) return;
    float beta = 1.f / (1.f + expf(-skip[0]));
    xout[t] = beta * o[t] + (1.f - beta) * xin[t];
}

// ---------------------------------------------------------------------------
// Edge attention / segment softmax (scatter phases, HBM-bound)
// ---------------------------------------------------------------------------
__global__ void edge_alpha_kernel(const float* __restrict__ q, const float* __restrict__ krel,
                                  const int* __restrict__ src, const int* __restrict__ dst,
                                  const float* __restrict__ p_rel, float* __restrict__ alpha,
                                  int E, float scale) {
    int t = blockIdx.x * blockDim.x + threadIdx.x;
    if (t >= E * HH) return;
    int e = t >> 2, h = t & 3;
    const float* qp = q    + (long)dst[e] * DD + h * DHD;
    const float* kp = krel + (long)src[e] * DD + h * DHD;
    float acc = 0.f;
#pragma unroll 4
    for (int i = 0; i < DHD; i += 4) {
        float4 qa = *(const float4*)(qp + i);
        float4 kb = *(const float4*)(kp + i);
        acc += qa.x * kb.x + qa.y * kb.y + qa.z * kb.z + qa.w * kb.w;
    }
    alpha[t] = acc * p_rel[h] * scale;
}

__global__ void seg_max_kernel(const float* __restrict__ alpha, const int* __restrict__ dst,
                               float* __restrict__ amax, int E) {
    int t = blockIdx.x * blockDim.x + threadIdx.x;
    if (t >= E * HH) return;
    int e = t >> 2, h = t & 3;
    atomicMaxF(&amax[(long)dst[e] * HH + h], alpha[t]);
}

__global__ void seg_exp_kernel(const float* __restrict__ alpha, const int* __restrict__ dst,
                               const float* __restrict__ amax, float* __restrict__ ex,
                               float* __restrict__ den, int E) {
    int t = blockIdx.x * blockDim.x + threadIdx.x;
    if (t >= E * HH) return;
    int e = t >> 2, h = t & 3;
    float am = amax[(long)dst[e] * HH + h];
    if (!(fabsf(am) <= 3.0e38f)) am = 0.f;   // isfinite guard, matches reference
    float v = expf(alpha[t] - am);
    ex[t] = v;
    atomicAdd(&den[(long)dst[e] * HH + h], v);
}

// one block (256 threads) per edge; thread c scatters channel c of the message
__global__ void seg_agg_kernel(const float* __restrict__ vrel, const float* __restrict__ ex,
                               const float* __restrict__ den, const int* __restrict__ src,
                               const int* __restrict__ dst, float* __restrict__ agg, int E) {
    int e = blockIdx.x;
    if (e >= E) return;
    int c = threadIdx.x;
    int h = c >> 6;
    int s = src[e], d = dst[e];
    float w = ex[(long)e * HH + h] / (den[(long)d * HH + h] + 1e-16f);
    atomicAdd(&agg[(long)d * DD + c], vrel[(long)s * DD + c] * w);
}

// ---------------------------------------------------------------------------
// Graph layernorm (PyG 'graph' mode): normalize over all nodes+channels per graph
// ---------------------------------------------------------------------------
__global__ void gln_stats_kernel(const float* __restrict__ x, const int* __restrict__ batch,
                                 float* __restrict__ gsum, float* __restrict__ gsq,
                                 float* __restrict__ cnt, int nrows) {
    int row = blockIdx.x;
    if (row >= nrows) return;
    int c = threadIdx.x;
    float v = x[(long)row * DD + c];
    __shared__ float s1[256], s2[256];
    s1[c] = v; s2[c] = v * v;
    __syncthreads();
    for (int s = 128; s > 0; s >>= 1) {
        if (c < s) { s1[c] += s1[c + s]; s2[c] += s2[c + s]; }
        __syncthreads();
    }
    if (c == 0) {
        int g = batch[row];
        atomicAdd(&gsum[g], s1[0]);
        atomicAdd(&gsq[g],  s2[0]);
        atomicAdd(&cnt[g],  1.f);
    }
}

__global__ void gln_apply_kernel(const float* __restrict__ x, const int* __restrict__ batch,
                                 const float* __restrict__ gsum, const float* __restrict__ gsq,
                                 const float* __restrict__ cnt, const float* __restrict__ w,
                                 const float* __restrict__ b, float* __restrict__ y, int nrows) {
    long t = (long)blockIdx.x * blockDim.x + threadIdx.x;
    if (t >= (long)nrows * DD) return;
    int row = (int)(t >> 8), c = (int)(t & 255);
    int g = batch[row];
    float denom = fmaxf(cnt[g], 1.f) * (float)DD;
    float mean  = gsum[g] / denom;
    float var   = gsq[g] / denom - mean * mean;
    y[t] = (x[t] - mean) * rsqrtf(var + 1e-5f) * w[c] + b[c];
}

// ---------------------------------------------------------------------------
// Pooling into padded g buffer [NB,GW]: per nt sum at `off`, max at `off+256`
// ---------------------------------------------------------------------------
__global__ void gbuf_init_kernel(float* gb) {
    int t = blockIdx.x * blockDim.x + threadIdx.x;
    if (t >= NB * GW) return;
    int c = t % GW;
    bool ismax = (c >= 256 && c < 512) || (c >= 768 && c < 1024);
    gb[t] = ismax ? -__builtin_inff() : 0.f;
}

__global__ void pool_kernel(const float* __restrict__ x, const int* __restrict__ batch,
                            float* __restrict__ gb, int nrows, int off) {
    long t = (long)blockIdx.x * blockDim.x + threadIdx.x;
    if (t >= (long)nrows * DD) return;
    int row = (int)(t >> 8), c = (int)(t & 255);
    int g = batch[row];
    float v = x[t];
    atomicAdd(&gb[g * GW + off + c], v);
    atomicMaxF(&gb[g * GW + off + 256 + c], v);
}

__global__ void gbuf_fix_kernel(float* gb) {
    int t = blockIdx.x * blockDim.x + threadIdx.x;
    if (t >= NB * GW) return;
    int c = t % GW;
    bool ismax = (c >= 256 && c < 512) || (c >= 768 && c < 1024);
    if (ismax && !(fabsf(gb[t]) <= 3.0e38f)) gb[t] = 0.f;
}

// y_base MLP: leaky_relu(y @ W1[1,16] + b1, 0.1) @ W2[16,16] + b2 -> gb[:,1024:1040]
__global__ void y_mlp_kernel(const float* __restrict__ yb, const float* __restrict__ W1,
                             const float* __restrict__ b1, const float* __restrict__ W2,
                             const float* __restrict__ b2, float* __restrict__ gb) {
    int t = blockIdx.x * blockDim.x + threadIdx.x;
    if (t >= NB * 16) return;
    int row = t >> 4, j = t & 15;
    float y = yb[row];
    float acc = b2[j];
    for (int k = 0; k < 16; ++k) {
        float h = y * W1[k] + b1[k];
        h = (h > 0.f) ? h : 0.1f * h;
        acc += h * W2[k * 16 + j];
    }
    gb[row * GW + 1024 + j] = acc;
}

// per-row layernorm (+optional GELU) for the final MLP, C <= 256
__global__ void rowln_kernel(float* __restrict__ x, const float* __restrict__ w,
                             const float* __restrict__ b, int C, int do_gelu) {
    int row = blockIdx.x;
    int c = threadIdx.x;
    __shared__ float sh[256];
    float v = (c < C) ? x[(long)row * C + c] : 0.f;
    sh[c] = v; __syncthreads();
    for (int s = 128; s > 0; s >>= 1) { if (c < s) sh[c] += sh[c + s]; __syncthreads(); }
    float mean = sh[0] / (float)C;
    __syncthreads();
    float d0 = (c < C) ? v - mean : 0.f;
    sh[c] = d0 * d0; __syncthreads();
    for (int s = 128; s > 0; s >>= 1) { if (c < s) sh[c] += sh[c + s]; __syncthreads(); }
    float var = sh[0] / (float)C;
    if (c < C) {
        float y = d0 * rsqrtf(var + 1e-5f) * w[c] + b[c];
        if (do_gelu) y = gelu_exact(y);
        x[(long)row * C + c] = y;
    }
}

// final head: out[row] = h2[row,:] @ Wg3[64,1] + bg3  (tiny, scalar)
__global__ void head_out_kernel(const float* __restrict__ h2, const float* __restrict__ Wg3,
                                const float* __restrict__ bg3, float* __restrict__ out) {
    int row = threadIdx.x;
    if (row >= NB) return;
    float acc = bg3[0];
    for (int k = 0; k < 64; ++k) acc += h2[row * 64 + k] * Wg3[k];
    out[row] = acc;
}

// ---------------------------------------------------------------------------
// Host orchestration
// ---------------------------------------------------------------------------
struct LinP { const float *Wk, *bk, *Wq, *bq, *Wv, *bv; };
struct RelP { const float *a, *m, *p; };
struct OutP { const float *Wa, *ba, *skip; };

extern "C" void kernel_launch(void* const* d_in, const int* in_sizes, int n_in,
                              void* d_out, int out_size, void* d_ws, size_t ws_size,
                              hipStream_t stream) {
    (void)in_sizes; (void)n_in; (void)out_size; (void)ws_size;

    // ---- input cursor (setup_inputs() insertion order, recursively) ----
    int ci = 0;
    auto F = [&]() { return (const float*)d_in[ci++]; };
    auto I = [&]() { return (const int*)d_in[ci++]; };

    const float* x_op  = F();
    const float* x_var = F();
    const float* x_dct[5]; for (int d = 0; d < 5; ++d) x_dct[d] = F();
    const float* y_base = F();

    // params.proj: nt order = [op, var, 5 directives]
    const float *projW[7], *projB[7];
    for (int i = 0; i < 7; ++i) { projW[i] = F(); projB[i] = F(); }

    auto getLin = [&]() { LinP l; l.Wk = F(); l.bk = F(); l.Wq = F(); l.bq = F(); l.Wv = F(); l.bv = F(); return l; };
    auto getRel = [&]() { RelP r; r.a = F(); r.m = F(); r.p = F(); return r; };
    auto getOut = [&]() { OutP o; o.Wa = F(); o.ba = F(); o.skip = F(); return o; };

    // params.hls: per directive {lin[d], lin[op], rel, out[d], out[op]}
    LinP hls_lin_d[5], hls_lin_op[5];
    RelP hls_rel[5];
    OutP hls_out_d[5], hls_out_op[5];
    for (int d = 0; d < 5; ++d) {
        hls_lin_d[d] = getLin(); hls_lin_op[d] = getLin();
        hls_rel[d]   = getRel();
        hls_out_d[d] = getOut(); hls_out_op[d] = getOut();
    }
    // params.conv: 4 layers {lin op, lin var, rel op/op, rel op/var, rel var/op, out op, out var}
    LinP cv_lin_op[4], cv_lin_var[4];
    RelP cv_r0[4], cv_r1[4], cv_r2[4];
    OutP cv_out_op[4], cv_out_var[4];
    for (int i = 0; i < 4; ++i) {
        cv_lin_op[i] = getLin(); cv_lin_var[i] = getLin();
        cv_r0[i] = getRel(); cv_r1[i] = getRel(); cv_r2[i] = getRel();
        cv_out_op[i] = getOut(); cv_out_var[i] = getOut();
    }
    // params.norm: 4 layers x {op:(w,b), var:(w,b)}
    const float *nw_op[4], *nb_op[4], *nw_var[4], *nb_var[4];
    for (int i = 0; i < 4; ++i) { nw_op[i] = F(); nb_op[i] = F(); nw_var[i] = F(); nb_var[i] = F(); }
    // params.y_mlp, params.g_mlp
    const float *yW1 = F(), *yb1 = F(), *yW2 = F(), *yb2 = F();
    const float *Wg1 = F(), *bg1 = F(), *g1w = F(), *g1b = F();
    const float *Wg2 = F(), *bg2 = F(), *g2w = F(), *g2b = F();
    const float *Wg3 = F(), *bg3 = F();

    // edge_index (each [2,E]: row0 = src, row1 = dst)
    const int* ei0 = I();  // op/to/op
    const int* ei1 = I();  // op/to/var
    const int* ei2 = I();  // var/to/op
    const int* batch_op  = I();
    const int* batch_var = I();
    const int* dct_idx[5];  for (int d = 0; d < 5; ++d) dct_idx[d]  = I();
    const int* dct_edge[5]; for (int d = 0; d < 5; ++d) dct_edge[d] = I();

    // ---- workspace bump allocator ----
    char* wp = (char*)d_ws;
    auto alloc = [&](long nf) { float* p = (float*)wp; wp += nf * sizeof(float); return p; };

    float* xopf  = alloc((long)N_OP * DD);
    float* xvarf = alloc((long)N_VAR * DD);
    float* xdctf[5]; for (int d = 0; d < 5; ++d) xdctf[d] = alloc((long)N_DCT * DD);
    float* nop   = alloc((long)N_OP * DD);
    float* nvar  = alloc((long)N_VAR * DD);
    float* k_op  = alloc((long)N_OP * DD);
    float* q_op  = alloc((long)N_OP * DD);
    float* v_op  = alloc((long)N_OP * DD);
    float* k_var = alloc((long)N_VAR * DD);
    float* q_var = alloc((long)N_VAR * DD);
    float* v_var = alloc((long)N_VAR * DD);
    float* krel  = alloc((long)N_OP * DD);   // relation K scratch (also o_var tmp)
    float* vrel0 = alloc((long)N_OP * DD);
    float* vrel1 = alloc((long)N_OP * DD);
    float* vrel2 = alloc((long)N_VAR * DD);
    float* agg_op  = alloc((long)N_OP * DD);
    float* agg_var = alloc((long)N_VAR * DD);
    float* tmp_o   = alloc((long)N_OP * DD); // o_op tmp; also gathered fxop in directive stage
    float* alphab  = alloc((long)(E_OPOP + E_OPVAR + E_VAROP) * HH);
    float* exb     = alloc((long)(E_OPOP + E_OPVAR + E_VAROP) * HH);
    float* amax_op  = alloc((long)N_OP * HH);
    float* den_op   = alloc((long)N_OP * HH);
    float* amax_var = alloc((long)N_VAR * HH);
    float* den_var  = alloc((long)N_VAR * HH);
    float* gbuf = alloc(NB * GW);
    float* Wg1p = alloc((long)GW * DD);      // zero-padded Wg1 (1056 x 256)
    float* h1   = alloc(NB * DD);
    float* h2   = alloc(NB * 64);
    float* cnt  = alloc(NB);
    float* gsum = alloc(NB);
    float* gsq  = alloc(NB);

    // ---- launch helpers ----
    auto gemm = [&](const float* A, const float* Bw, const float* bias, float* C,
                    int M, int N, int K, int lda, int ldb, int ldc,
                    long aB, long bB, long cB, int nb, int act) {
        dim3 g(N / 64, M / 16, nb);
        wmma_gemm_kernel<<<g, dim3(32), 0, stream>>>(A, Bw, bias, C, K,
                                                     lda, ldb, ldc, aB, bB, cB, act);
    };
    auto gemm1 = [&](const float* A, const float* Bw, const float* bias, float* C,
                     int M, int N, int K, int act) {
        gemm(A, Bw, bias, C, M, N, K, K, N, N, 0, 0, 0, 1, act);
    };
    // per-head relation transform: C[n,h,:] = A[n,h,:] @ R[h]  (batched over heads)
    auto rel_gemm = [&](const float* A, const float* R, float* C, int n) {
        gemm(A, R, nullptr, C, n, DHD, DHD, DD, DHD, DD,
             (long)DHD, (long)DHD * DHD, (long)DHD, HH, 0);
    };
    auto fill = [&](float* p, float v, long n) {
        fill_kernel<<<dim3((unsigned)((n + 255) / 256)), dim3(256), 0, stream>>>(p, v, n);
    };
    auto ew_grid = [](long n) { return dim3((unsigned)((n + 255) / 256)); };
    const float NEG_INF = -__builtin_inff();
    const float scale = 0.125f; // 1/sqrt(64)

    auto seg_pass = [&](const float* alpha, const int* dst, float* amax, int E) {
        seg_max_kernel<<<ew_grid((long)E * HH), dim3(256), 0, stream>>>(alpha, dst, amax, E);
    };
    auto exp_pass = [&](const float* alpha, float* ex, const int* dst,
                        const float* amax, float* den, int E) {
        seg_exp_kernel<<<ew_grid((long)E * HH), dim3(256), 0, stream>>>(alpha, dst, amax, ex, den, E);
    };
    auto agg_pass = [&](const float* vrel, const float* ex, const float* den,
                        const int* src, const int* dst, float* agg, int E) {
        seg_agg_kernel<<<dim3(E), dim3(256), 0, stream>>>(vrel, ex, den, src, dst, agg, E);
    };

    // ======================= 1) input projections (WMMA) =======================
    gemm1(x_op,  projW[0], projB[0], xopf,  N_OP,  DD, 64, 0);
    gemm1(x_var, projW[1], projB[1], xvarf, N_VAR, DD, 64, 0);
    for (int d = 0; d < 5; ++d)
        gemm1(x_dct[d], projW[2 + d], projB[2 + d], xdctf[d], N_DCT, DD, 64, 0);

    // ======================= 2) directive HGT convs (sequential) ==============
    for (int d = 0; d < 5; ++d) {
        const int* esrc = dct_edge[d];
        const int* edst = dct_edge[d] + E_DCTE;
        float* fxop  = tmp_o;   // gathered op features [10000,256]
        float* kd    = k_var;   // reuse big buffers
        float* vd    = v_var;
        float* qog   = q_op;
        float* kreld = krel;
        float* vreld = vrel2;
        float* o10k  = vrel0;

        gather_rows_kernel<<<ew_grid((long)M_OPG * DD), dim3(256), 0, stream>>>(xopf, dct_idx[d], fxop, M_OPG);
        gemm1(xdctf[d], hls_lin_d[d].Wk, hls_lin_d[d].bk, kd,  N_DCT, DD, DD, 0);
        gemm1(xdctf[d], hls_lin_d[d].Wv, hls_lin_d[d].bv, vd,  N_DCT, DD, DD, 0);
        gemm1(fxop,     hls_lin_op[d].Wq, hls_lin_op[d].bq, qog, M_OPG, DD, DD, 0);
        rel_gemm(kd, hls_rel[d].a, kreld, N_DCT);
        rel_gemm(vd, hls_rel[d].m, vreld, N_DCT);

        edge_alpha_kernel<<<ew_grid((long)E_DCTE * HH), dim3(256), 0, stream>>>(
            qog, kreld, esrc, edst, hls_rel[d].p, alphab, E_DCTE, scale);

        fill(amax_op, NEG_INF, (long)M_OPG * HH);
        fill(den_op,  0.f,     (long)M_OPG * HH);
        fill(agg_op,  0.f,     (long)M_OPG * DD);
        seg_pass(alphab, edst, amax_op, E_DCTE);
        exp_pass(alphab, exb, edst, amax_op, den_op, E_DCTE);
        agg_pass(vreld, exb, den_op, esrc, edst, agg_op, E_DCTE);

        gelu_kernel<<<ew_grid((long)M_OPG * DD), dim3(256), 0, stream>>>(agg_op, (long)M_OPG * DD);
        gemm1(agg_op, hls_out_op[d].Wa, hls_out_op[d].ba, o10k, M_OPG, DD, DD, 0);
        blend_kernel<<<ew_grid((long)M_OPG * DD), dim3(256), 0, stream>>>(
            o10k, fxop, hls_out_op[d].skip, fxop, (long)M_OPG * DD);
        scatter_rows_kernel<<<ew_grid((long)M_OPG * DD), dim3(256), 0, stream>>>(fxop, dct_idx[d], xopf, M_OPG);
    }

    // ======================= 3) main HGT layers ================================
    const int* s0 = ei0;            const int* d0 = ei0 + E_OPOP;
    const int* s1 = ei1;            const int* d1 = ei1 + E_OPVAR;
    const int* s2 = ei2;            const int* d2 = ei2 + E_VAROP;
    float* a0 = alphab;                         float* e0 = exb;
    float* a1 = alphab + (long)E_OPOP * HH;     float* e1 = exb + (long)E_OPOP * HH;
    float* a2 = a1 + (long)E_OPVAR * HH;        float* e2 = e1 + (long)E_OPVAR * HH;

    for (int L = 0; L < 4; ++L) {
        // graph layernorm (per node type)
        fill(cnt, 0.f, NB); fill(gsum, 0.f, NB); fill(gsq, 0.f, NB);
        gln_stats_kernel<<<dim3(N_OP), dim3(256), 0, stream>>>(xopf, batch_op, gsum, gsq, cnt, N_OP);
        gln_apply_kernel<<<ew_grid((long)N_OP * DD), dim3(256), 0, stream>>>(
            xopf, batch_op, gsum, gsq, cnt, nw_op[L], nb_op[L], nop, N_OP);
        fill(cnt, 0.f, NB); fill(gsum, 0.f, NB); fill(gsq, 0.f, NB);
        gln_stats_kernel<<<dim3(N_VAR), dim3(256), 0, stream>>>(xvarf, batch_var, gsum, gsq, cnt, N_VAR);
        gln_apply_kernel<<<ew_grid((long)N_VAR * DD), dim3(256), 0, stream>>>(
            xvarf, batch_var, gsum, gsq, cnt, nw_var[L], nb_var[L], nvar, N_VAR);

        // K/Q/V projections (6 WMMA GEMMs)
        gemm1(nop,  cv_lin_op[L].Wk,  cv_lin_op[L].bk,  k_op,  N_OP,  DD, DD, 0);
        gemm1(nop,  cv_lin_op[L].Wq,  cv_lin_op[L].bq,  q_op,  N_OP,  DD, DD, 0);
        gemm1(nop,  cv_lin_op[L].Wv,  cv_lin_op[L].bv,  v_op,  N_OP,  DD, DD, 0);
        gemm1(nvar, cv_lin_var[L].Wk, cv_lin_var[L].bk, k_var, N_VAR, DD, DD, 0);
        gemm1(nvar, cv_lin_var[L].Wq, cv_lin_var[L].bq, q_var, N_VAR, DD, DD, 0);
        gemm1(nvar, cv_lin_var[L].Wv, cv_lin_var[L].bv, v_var, N_VAR, DD, DD, 0);

        // relation transforms + edge attention logits
        rel_gemm(k_op, cv_r0[L].a, krel, N_OP);
        rel_gemm(v_op, cv_r0[L].m, vrel0, N_OP);
        edge_alpha_kernel<<<ew_grid((long)E_OPOP * HH), dim3(256), 0, stream>>>(
            q_op, krel, s0, d0, cv_r0[L].p, a0, E_OPOP, scale);

        rel_gemm(k_op, cv_r1[L].a, krel, N_OP);
        rel_gemm(v_op, cv_r1[L].m, vrel1, N_OP);
        edge_alpha_kernel<<<ew_grid((long)E_OPVAR * HH), dim3(256), 0, stream>>>(
            q_var, krel, s1, d1, cv_r1[L].p, a1, E_OPVAR, scale);

        rel_gemm(k_var, cv_r2[L].a, krel, N_VAR);
        rel_gemm(v_var, cv_r2[L].m, vrel2, N_VAR);
        edge_alpha_kernel<<<ew_grid((long)E_VAROP * HH), dim3(256), 0, stream>>>(
            q_op, krel, s2, d2, cv_r2[L].p, a2, E_VAROP, scale);

        // segment softmax + aggregation, dst type = op (relations 0 and 2)
        fill(amax_op, NEG_INF, (long)N_OP * HH);
        fill(den_op,  0.f,     (long)N_OP * HH);
        fill(agg_op,  0.f,     (long)N_OP * DD);
        seg_pass(a0, d0, amax_op, E_OPOP);
        seg_pass(a2, d2, amax_op, E_VAROP);
        exp_pass(a0, e0, d0, amax_op, den_op, E_OPOP);
        exp_pass(a2, e2, d2, amax_op, den_op, E_VAROP);
        agg_pass(vrel0, e0, den_op, s0, d0, agg_op, E_OPOP);
        agg_pass(vrel2, e2, den_op, s2, d2, agg_op, E_VAROP);

        // dst type = var (relation 1)
        fill(amax_var, NEG_INF, (long)N_VAR * HH);
        fill(den_var,  0.f,     (long)N_VAR * HH);
        fill(agg_var,  0.f,     (long)N_VAR * DD);
        seg_pass(a1, d1, amax_var, E_OPVAR);
        exp_pass(a1, e1, d1, amax_var, den_var, E_OPVAR);
        agg_pass(vrel1, e1, den_var, s1, d1, agg_var, E_OPVAR);

        // output projection + gated skip
        gelu_kernel<<<ew_grid((long)N_OP * DD), dim3(256), 0, stream>>>(agg_op, (long)N_OP * DD);
        gemm1(agg_op, cv_out_op[L].Wa, cv_out_op[L].ba, tmp_o, N_OP, DD, DD, 0);
        blend_kernel<<<ew_grid((long)N_OP * DD), dim3(256), 0, stream>>>(
            tmp_o, nop, cv_out_op[L].skip, xopf, (long)N_OP * DD);

        gelu_kernel<<<ew_grid((long)N_VAR * DD), dim3(256), 0, stream>>>(agg_var, (long)N_VAR * DD);
        gemm1(agg_var, cv_out_var[L].Wa, cv_out_var[L].ba, krel, N_VAR, DD, DD, 0);
        blend_kernel<<<ew_grid((long)N_VAR * DD), dim3(256), 0, stream>>>(
            krel, nvar, cv_out_var[L].skip, xvarf, (long)N_VAR * DD);
    }

    // ======================= 4) pooling + head ================================
    gbuf_init_kernel<<<ew_grid(NB * GW), dim3(256), 0, stream>>>(gbuf);
    pool_kernel<<<ew_grid((long)N_OP * DD), dim3(256), 0, stream>>>(xopf, batch_op, gbuf, N_OP, 0);
    pool_kernel<<<ew_grid((long)N_VAR * DD), dim3(256), 0, stream>>>(xvarf, batch_var, gbuf, N_VAR, 512);
    gbuf_fix_kernel<<<ew_grid(NB * GW), dim3(256), 0, stream>>>(gbuf);
    y_mlp_kernel<<<dim3(1), dim3(256), 0, stream>>>(y_base, yW1, yb1, yW2, yb2, gbuf);

    // zero-padded Wg1 (rows 1040..1055 = 0) so head GEMM1 runs the fast path
    fill(Wg1p, 0.f, (long)GW * DD);
    copy_kernel<<<ew_grid(1040L * DD), dim3(256), 0, stream>>>(Wg1, Wg1p, 1040L * DD);

    gemm1(gbuf, Wg1p, bg1, h1, NB, DD, GW, 0);
    rowln_kernel<<<dim3(NB), dim3(256), 0, stream>>>(h1, g1w, g1b, DD, 1);
    gemm1(h1, Wg2, bg2, h2, NB, 64, DD, 0);
    rowln_kernel<<<dim3(NB), dim3(256), 0, stream>>>(h2, g2w, g2b, 64, 1);
    head_out_kernel<<<dim3(1), dim3(32), 0, stream>>>(h2, Wg3, bg3, (float*)d_out);
}
